// DrowsinessLSTM_24756191494614
// MI455X (gfx1250) — matrix operand order repaired and checked by reading, not compiled, not run on hardware
//
#include <hip/hip_runtime.h>

typedef __attribute__((ext_vector_type(16))) _Float16 v16h;
typedef __attribute__((ext_vector_type(8)))  float    v8f;

namespace {

constexpr int T_LEN  = 512;
constexpr int IN_DIM = 2;
constexpr int HDIM   = 64;
constexpr int ROWS   = 16;     // batch rows per workgroup (WMMA M)
constexpr int NWAVE  = 4;      // gate-column split: wave w owns tiles {w,4+w,8+w,12+w}

// A-operand staging buffers (halves): [kc][half][m][e], double buffered
constexpr int HX0_H = 3 * 2 * 16 * 16;   // layer0 A: K = 0..95 (h0 | x | zero-pad)
constexpr int HX1_H = 4 * 2 * 16 * 16;   // layer1 A: K = 0..127 (h1 | h0_t)

// gfx1250 hardware V_TANH_F32 (TRANS op); exp fallback keeps it compile-safe.
__device__ __forceinline__ float fast_tanh(float x) {
#if __has_builtin(__builtin_amdgcn_tanhf)
  return __builtin_amdgcn_tanhf(x);
#elif __has_builtin(__builtin_amdgcn_tanh_f32)
  return __builtin_amdgcn_tanh_f32(x);
#else
  return 2.0f / (1.0f + __expf(-2.0f * x)) - 1.0f;
#endif
}

// sigmoid(x + b) with hb = 0.5*b prefolded: 0.5 + 0.5*tanh(0.5*x + hb)
__device__ __forceinline__ float sigm_b(float x, float hb) {
  return __builtin_fmaf(fast_tanh(__builtin_fmaf(x, 0.5f, hb)), 0.5f, 0.5f);
}

// Index (halves) of (row m, column K) in an A staging buffer, matching the
// contiguous k order used for both A and B in this kernel:
//   lanes 0-15 : e=0..15 -> K=0..15
//   lanes 16-31: e=0..15 -> K=16..31
__device__ __forceinline__ int aidx(int m, int K) {
  int kc  = K >> 5;
  int rem = K & 31;
  int hh  = rem >> 4;
  int e   = rem & 15;
  return ((kc * 2 + hh) * 16 + m) * 16 + e;
}

} // namespace

__global__ __launch_bounds__(NWAVE * 32)   // 128 threads = 4 waves, one WGP
void lstm2_fused(const float* __restrict__ x,
                 const float* __restrict__ W_ih0, const float* __restrict__ W_hh0,
                 const float* __restrict__ b_ih0, const float* __restrict__ b_hh0,
                 const float* __restrict__ W_ih1, const float* __restrict__ W_hh1,
                 const float* __restrict__ b_ih1, const float* __restrict__ b_hh1,
                 const float* __restrict__ W_fc,  const float* __restrict__ b_fc,
                 float* __restrict__ out)
{
  __shared__ _Float16 HX0[2 * HX0_H];      // ping-pong layer-0 A
  __shared__ _Float16 HX1[2 * HX1_H];      // ping-pong layer-1 A
  __shared__ float    h1fin[ROWS * HDIM];  // final h1 for the FC head

  const int tid  = threadIdx.x;
  const int wv   = tid >> 5;               // gate-column group of this wave
  const int lane = tid & 31;
  const int m0   = lane & 15;              // A row / D column within tile
  const int hh2  = lane >> 4;              // lane half
  const int mb   = 8 * hh2;                // D row base for this lane
  const int Rrow = blockIdx.x * ROWS;

  // zero both A buffers once (also provides layer-0 zero pad K = 66..95)
  for (int i = tid; i < 2 * HX0_H; i += NWAVE * 32) HX0[i] = (_Float16)0.0f;
  for (int i = tid; i < 2 * HX1_H; i += NWAVE * 32) HX1[i] = (_Float16)0.0f;

  // ---- register-resident, loop-invariant B operands (224 VGPRs / wave) ----
  // element e of lane l == (K = 32*kc + 16*(l>>4) + e, N = 16*tile + (l&15))
  v16h B0[3][4];   // layer0: [kc][gate], tile = 4*gate + wv, K = [h0 | x | 0]
  v16h B1[4][4];   // layer1: [kc][gate],                     K = [h1 | h0_t]
  #pragma unroll
  for (int kc = 0; kc < 3; ++kc)
    #pragma unroll
    for (int g = 0; g < 4; ++g) {
      const int N = 16 * (4 * g + wv) + m0;
      #pragma unroll
      for (int e = 0; e < 16; ++e) {
        const int K = 32 * kc + 16 * hh2 + e;
        float v = 0.0f;
        if (K < 64)      v = W_hh0[N * 64 + K];
        else if (K < 66) v = W_ih0[N * 2 + (K - 64)];
        B0[kc][g][e] = (_Float16)v;
      }
    }
  #pragma unroll
  for (int kc = 0; kc < 4; ++kc)
    #pragma unroll
    for (int g = 0; g < 4; ++g) {
      const int N = 16 * (4 * g + wv) + m0;
      #pragma unroll
      for (int e = 0; e < 16; ++e) {
        const int K = 32 * kc + 16 * hh2 + e;
        B1[kc][g][e] = (_Float16)((K < 64) ? W_hh1[N * 64 + K]
                                           : W_ih1[N * 64 + (K - 64)]);
      }
    }

  // per-lane gate column j and prefolded biases (loop-invariant scalars)
  const int j = 16 * wv + m0;
  const float bi0 = 0.5f * (b_ih0[j]       + b_hh0[j]);
  const float bf0 = 0.5f * (b_ih0[64 + j]  + b_hh0[64 + j]);
  const float bg0 =        (b_ih0[128 + j] + b_hh0[128 + j]);
  const float bo0 = 0.5f * (b_ih0[192 + j] + b_hh0[192 + j]);
  const float bi1 = 0.5f * (b_ih1[j]       + b_hh1[j]);
  const float bf1 = 0.5f * (b_ih1[64 + j]  + b_hh1[64 + j]);
  const float bg1 =        (b_ih1[128 + j] + b_hh1[128 + j]);
  const float bo1 = 0.5f * (b_ih1[192 + j] + b_hh1[192 + j]);

  float c0[8], c1[8];
  #pragma unroll
  for (int r = 0; r < 8; ++r) { c0[r] = 0.0f; c1[r] = 0.0f; }

  // wave 0 stages x: lane (m0, hh2) streams x[Rrow+m0][t][hh2]
  const float* xp = x + (size_t)(Rrow + m0) * T_LEN * IN_DIM + hh2;
  float xcur = (wv == 0) ? xp[0] : 0.0f;

  __syncthreads();   // LDS zero-init visible to everyone

  for (int t = 0; t < T_LEN; ++t) {
    const int cur = t & 1;
    _Float16* A0c = HX0 + cur * HX0_H;          // layer0 A read this step
    _Float16* A0n = HX0 + (cur ^ 1) * HX0_H;    // h0 written for next step
    _Float16* A1c = HX1 + cur * HX1_H;          // layer1 A read this step
    _Float16* A1n = HX1 + (cur ^ 1) * HX1_H;    // h1 written for next step

    if (wv == 0) {                               // stage x_t at K = 64 + hh2
      A0c[(4 * 16 + m0) * 16 + hh2] = (_Float16)xcur;
      if (t + 1 < T_LEN) xcur = xp[(size_t)(t + 1) * IN_DIM];
      if (t + 8 < T_LEN) __builtin_prefetch(xp + (size_t)(t + 8) * IN_DIM, 0, 1);
    }
    __syncthreads();   // x_t staged; everyone past previous step's stores

    // ---- layer 0: gates = [h0 | x | 0] x Wc0 (K = 96), C starts at 0 ----
    v16h a0[3];
    #pragma unroll
    for (int kc = 0; kc < 3; ++kc)
      a0[kc] = *(const v16h*)(A0c + ((kc * 2 + hh2) * 16 + m0) * 16);

    {
      v8f acc[4] = {};
      #pragma unroll
      for (int kc = 0; kc < 3; ++kc)
        #pragma unroll
        for (int g = 0; g < 4; ++g)
          acc[g] = __builtin_amdgcn_wmma_f32_16x16x32_f16(
              false, a0[kc], false, B0[kc][g], (short)0, acc[g], false, false);

      #pragma unroll
      for (int r = 0; r < 8; ++r) {             // lane-local LSTM cell update
        float ig = sigm_b(acc[0][r], bi0);
        float fg = sigm_b(acc[1][r], bf0);
        float gg = fast_tanh(acc[2][r] + bg0);
        float og = sigm_b(acc[3][r], bo0);
        float c  = __builtin_fmaf(fg, c0[r], ig * gg);
        c0[r]    = c;
        float h  = og * fast_tanh(c);
        int   m  = mb + r;
        _Float16 hf = (_Float16)h;
        A0n[aidx(m, j)]      = hf;              // next step's layer-0 A
        A1c[aidx(m, 64 + j)] = hf;              // this step's layer-1 A (h0_t)
      }
    }
    __syncthreads();   // all waves' h0_t in place before layer-1 A loads

    // ---- layer 1: gates = [h1 | h0_t] x Wc1 (K = 128), C starts at 0 ----
    v16h a1[4];
    #pragma unroll
    for (int kc = 0; kc < 4; ++kc)
      a1[kc] = *(const v16h*)(A1c + ((kc * 2 + hh2) * 16 + m0) * 16);

    {
      v8f acc[4] = {};
      #pragma unroll
      for (int kc = 0; kc < 4; ++kc)
        #pragma unroll
        for (int g = 0; g < 4; ++g)
          acc[g] = __builtin_amdgcn_wmma_f32_16x16x32_f16(
              false, a1[kc], false, B1[kc][g], (short)0, acc[g], false, false);

      const bool last = (t == T_LEN - 1);
      #pragma unroll
      for (int r = 0; r < 8; ++r) {
        float ig = sigm_b(acc[0][r], bi1);
        float fg = sigm_b(acc[1][r], bf1);
        float gg = fast_tanh(acc[2][r] + bg1);
        float og = sigm_b(acc[3][r], bo1);
        float c  = __builtin_fmaf(fg, c1[r], ig * gg);
        c1[r]    = c;
        float h  = og * fast_tanh(c);
        int   m  = mb + r;
        A1n[aidx(m, j)] = (_Float16)h;          // next step's layer-1 A (h1)
        if (last) h1fin[m * HDIM + j] = h;
      }
    }
  }

  __syncthreads();

  // ------------------------- FC head + sigmoid -------------------------
  if (wv == 0 && lane < 16) {
    const float* hr = h1fin + lane * HDIM;
    float s = b_fc[0];
    #pragma unroll 8
    for (int jj = 0; jj < HDIM; ++jj) s += hr[jj] * W_fc[jj];
    out[Rrow + lane] = __builtin_fmaf(fast_tanh(0.5f * s), 0.5f, 0.5f);
  }
}

extern "C" void kernel_launch(void* const* d_in, const int* in_sizes, int n_in,
                              void* d_out, int out_size, void* d_ws, size_t ws_size,
                              hipStream_t stream) {
  const float* x     = (const float*)d_in[0];
  const float* W_ih0 = (const float*)d_in[1];
  const float* W_hh0 = (const float*)d_in[2];
  const float* b_ih0 = (const float*)d_in[3];
  const float* b_hh0 = (const float*)d_in[4];
  const float* W_ih1 = (const float*)d_in[5];
  const float* W_hh1 = (const float*)d_in[6];
  const float* b_ih1 = (const float*)d_in[7];
  const float* b_hh1 = (const float*)d_in[8];
  const float* W_fc  = (const float*)d_in[9];
  const float* b_fc  = (const float*)d_in[10];
  float* out = (float*)d_out;

  const int Bn = in_sizes[0] / (T_LEN * IN_DIM);   // 4096
  dim3 grid(Bn / ROWS);                             // 256 persistent workgroups
  dim3 block(NWAVE * 32);                           // 4 waves on one WGP

  hipLaunchKernelGGL(lstm2_fused, grid, block, 0, stream,
                     x, W_ih0, W_hh0, b_ih0, b_hh0,
                     W_ih1, W_hh1, b_ih1, b_hh1, W_fc, b_fc, out);
}
